// AttentionBlock_37512244363966
// MI455X (gfx1250) — compile-verified, matrix operand order checked by reading, not run on hardware
//
#include <hip/hip_runtime.h>
#include <hip/hip_bf16.h>
#include <math.h>

typedef __attribute__((ext_vector_type(16))) __bf16 v16bf;
typedef __attribute__((ext_vector_type(8)))  float  v8f;

#define TID (threadIdx.x)

// ---- problem constants (B=4, C=512, H=W=64)
#define BATCH   4
#define CCH     512
#define HW      4096
#define CQKV    1536
#define NGROUPS 32
#define GSIZE   16
#define GN_N    (GSIZE * HW)
#define EPS     1e-5f
#define QSCALE  0.04419417382415922f  // 1/sqrt(512), folded into Q

// ---- GEMM tiling: 128x128 block tile, K-step 32, 8 wave32s in 4(M)x2(N)
// Each wave owns a 32x64 sub-tile = 2x4 WMMA accumulators.
#define BM 128
#define BN 128
#define BK 32
#define LDST 40              // padded LDS row stride: 80B = 16B-aligned, 20 banks
#define ABUF (BM * LDST)
#define BBUF (BN * LDST)

#define WMMA_BF16(a, b, c)                                                    \
  __builtin_amdgcn_wmma_f32_16x16x32_bf16(false, (a), false, (b), (short)0,   \
                                          (c), false, false)

// ------------------------------------------------------------------
// CDNA5 async copy: global -> LDS, 16B per lane, tracked by ASYNCcnt.
// 128 rows x 32 cols (row-major, stride ld) -> LDS stride LDST.
// ------------------------------------------------------------------
__device__ __forceinline__ void stage_async(const __bf16* src, int ld,
                                            __bf16* dst) {
#pragma unroll
  for (int i = 0; i < 2; ++i) {
    int idx = TID + 256 * i;
    int r = idx >> 2, c = (idx & 3) * 8;
    const __bf16* g = src + (size_t)r * ld + c;
    unsigned l = (unsigned)(uintptr_t)(dst + r * LDST + c);  // LDS byte offset
    asm volatile("global_load_async_to_lds_b128 %0, %1, off"
                 :: "v"(l), "v"(g) : "memory");
  }
}
__device__ __forceinline__ void wait_async() {
  asm volatile("s_wait_asynccnt 0x0" ::: "memory");
}

// Transpose staging (register round-trip; async copies cannot transpose):
// global tile 32(K) x 128(N) K-major -> LDS [n][k].
struct StageT { uint4 v[2]; };

__device__ __forceinline__ void stageT_load(StageT& s, const __bf16* src, int ld) {
#pragma unroll
  for (int i = 0; i < 2; ++i) {
    int idx = TID + 256 * i;
    int k = idx >> 4, nc = (idx & 15) * 8;
    s.v[i] = *(const uint4*)(src + (size_t)k * ld + nc);
  }
}
__device__ __forceinline__ void stageT_store(const StageT& s, __bf16* dst) {
#pragma unroll
  for (int i = 0; i < 2; ++i) {
    int idx = TID + 256 * i;
    int k = idx >> 4, nc = (idx & 15) * 8;
    const __bf16* e = (const __bf16*)&s.v[i];
#pragma unroll
    for (int j = 0; j < 8; ++j) dst[(nc + j) * LDST + k] = e[j];
  }
}

// ------------------------------------------------------------------
// WMMA fragment: 16(M/N) x 32(K) bf16 from LDS (stride LDST).
// lanes 0-15: row = lane, K 0..7 & 16..23; lanes 16-31: row = lane-16,
// K 8..15 & 24..31  -> two contiguous 16B chunks = 2 x ds_load_b128.
// ------------------------------------------------------------------
__device__ __forceinline__ v16bf load_frag(const __bf16* base) {
  const int lane = TID & 31;
  const __bf16* p = base + (lane & 15) * LDST + ((lane & 16) >> 1);
  v16bf f;
#pragma unroll
  for (int e = 0; e < 8; ++e) f[e] = p[e];
#pragma unroll
  for (int e = 0; e < 8; ++e) f[e + 8] = p[e + 16];
  return f;
}

struct Acc { v8f t[2][4]; };

// Double-buffered pipelined GEMM (one barrier per K-step):
//   async-copy tile k+1 -> WMMA on tile k -> s_wait_asynccnt -> barrier.
template <bool TRANSB>
__device__ __forceinline__ void gemm_tile(const __bf16* A, int lda,
                                          const __bf16* B, int ldb, int kdim,
                                          __bf16* ldsA, __bf16* ldsB, Acc& acc) {
  const int w  = TID >> 5;
  const int wm = (w >> 1) * 32;
  const int wn = (w & 1) * 64;
  v8f zero = {};
#pragma unroll
  for (int i = 0; i < 2; ++i)
#pragma unroll
    for (int j = 0; j < 4; ++j) acc.t[i][j] = zero;

  StageT sbt;
  stage_async(A, lda, ldsA);
  if constexpr (TRANSB) { stageT_load(sbt, B, ldb); stageT_store(sbt, ldsB); }
  else                  { stage_async(B, ldb, ldsB); }
  wait_async();
  __syncthreads();

  const int nk = kdim / BK;
  for (int k = 0; k < nk; ++k) {
    const int cur = k & 1;
    if (k + 1 < nk) {  // start filling the other buffer right away
      stage_async(A + (size_t)(k + 1) * BK, lda, ldsA + (1 - cur) * ABUF);
      if constexpr (TRANSB)
        stageT_load(sbt, B + (size_t)(k + 1) * BK * ldb, ldb);
      else
        stage_async(B + (size_t)(k + 1) * BK, ldb, ldsB + (1 - cur) * BBUF);
    }
    const __bf16* cA = ldsA + cur * ABUF;
    const __bf16* cB = ldsB + cur * BBUF;
    v16bf a0 = load_frag(cA + (wm +  0) * LDST);
    v16bf a1 = load_frag(cA + (wm + 16) * LDST);
    v16bf b0 = load_frag(cB + (wn +  0) * LDST);
    v16bf b1 = load_frag(cB + (wn + 16) * LDST);
    v16bf b2 = load_frag(cB + (wn + 32) * LDST);
    v16bf b3 = load_frag(cB + (wn + 48) * LDST);
    acc.t[0][0] = WMMA_BF16(a0, b0, acc.t[0][0]);
    acc.t[0][1] = WMMA_BF16(a0, b1, acc.t[0][1]);
    acc.t[0][2] = WMMA_BF16(a0, b2, acc.t[0][2]);
    acc.t[0][3] = WMMA_BF16(a0, b3, acc.t[0][3]);
    acc.t[1][0] = WMMA_BF16(a1, b0, acc.t[1][0]);
    acc.t[1][1] = WMMA_BF16(a1, b1, acc.t[1][1]);
    acc.t[1][2] = WMMA_BF16(a1, b2, acc.t[1][2]);
    acc.t[1][3] = WMMA_BF16(a1, b3, acc.t[1][3]);
    if (k + 1 < nk) {
      if constexpr (TRANSB)
        stageT_store(sbt, ldsB + (1 - cur) * BBUF);
      wait_async();  // async tile k+1 landed (latency hidden by WMMAs)
    }
    __syncthreads();
  }
}

// Accumulator element -> (row, col) within the 128x128 block tile.
#define EPILOGUE_POS()                                                       \
  const int w = TID >> 5, wm = (w >> 1) * 32, wn = (w & 1) * 64;             \
  const int lane = TID & 31, colb = lane & 15, rowb = (lane & 16) >> 1;

// ------------------------------------------------------------------
// Kernels
// ------------------------------------------------------------------
__global__ void __launch_bounds__(256)
convert_w_kernel(const float* qkv_w, const float* out_w,
                 __bf16* qkv_wb, __bf16* out_wb) {
  int i = blockIdx.x * 256 + TID;
  if (i < CQKV * CCH) qkv_wb[i] = (__bf16)qkv_w[i];
  if (i < CCH * CCH)  out_wb[i] = (__bf16)out_w[i];
}

__global__ void __launch_bounds__(256)
groupnorm_kernel(const float* x, const float* w, const float* b,
                 float* xn, __bf16* xnb) {
  const int z = blockIdx.x >> 5;
  const int g = blockIdx.x & 31;
  const size_t base = ((size_t)z * CCH + (size_t)g * GSIZE) * HW;
  __shared__ float s_sum[256], s_sq[256];
  float sum = 0.f, sq = 0.f;
  for (int i = TID; i < GN_N; i += 256) {
    float v = x[base + i];
    sum += v; sq += v * v;
  }
  s_sum[TID] = sum; s_sq[TID] = sq;
  __syncthreads();
  for (int s = 128; s > 0; s >>= 1) {
    if (TID < s) { s_sum[TID] += s_sum[TID + s]; s_sq[TID] += s_sq[TID + s]; }
    __syncthreads();
  }
  const float mean = s_sum[0] * (1.0f / GN_N);
  const float var  = s_sq[0] * (1.0f / GN_N) - mean * mean;
  const float rinv = rsqrtf(var + EPS);
  for (int i = TID; i < GN_N; i += 256) {
    int c = g * GSIZE + (i >> 12);
    float v = (x[base + i] - mean) * rinv * w[c] + b[c];
    xn[base + i]  = v;
    xnb[base + i] = (__bf16)v;
  }
}

// qkv = W(1536x512) * xn(512x4096) + b ; also emit pre-scaled Q^T (4096x512)
__global__ void __launch_bounds__(256)
gemm_qkv_kernel(const __bf16* Wq, const __bf16* xn, const float* bias,
                __bf16* qkv, __bf16* qT) {
  __shared__ __bf16 ldsA[2 * ABUF];
  __shared__ __bf16 ldsB[2 * BBUF];
  const int z = blockIdx.z, m0 = blockIdx.y * BM, n0 = blockIdx.x * BN;
  Acc acc;
  gemm_tile<true>(Wq + (size_t)m0 * CCH, CCH,
                  xn + (size_t)z * CCH * HW + n0, HW, CCH, ldsA, ldsB, acc);
  EPILOGUE_POS();
#pragma unroll
  for (int mi = 0; mi < 2; ++mi)
#pragma unroll
    for (int ni = 0; ni < 4; ++ni)
#pragma unroll
      for (int r = 0; r < 8; ++r) {
        int grow = m0 + wm + mi * 16 + rowb + r;
        int gcol = n0 + wn + ni * 16 + colb;
        float v = acc.t[mi][ni][r] + bias[grow];
        qkv[((size_t)z * CQKV + grow) * HW + gcol] = (__bf16)v;
        if (grow < CCH)
          qT[((size_t)z * HW + gcol) * CCH + grow] = (__bf16)(v * QSCALE);
      }
}

// S(4096x4096) = qT(4096x512) * K(512x4096), fp32 out (scale folded into Q)
__global__ void __launch_bounds__(256)
gemm_scores_kernel(const __bf16* qT, const __bf16* kmat, float* scores) {
  __shared__ __bf16 ldsA[2 * ABUF];
  __shared__ __bf16 ldsB[2 * BBUF];
  const int z = blockIdx.z, m0 = blockIdx.y * BM, n0 = blockIdx.x * BN;
  Acc acc;
  gemm_tile<true>(qT + ((size_t)z * HW + m0) * CCH, CCH,
                  kmat + (size_t)z * CQKV * HW + (size_t)CCH * HW + n0, HW,
                  CCH, ldsA, ldsB, acc);
  EPILOGUE_POS();
#pragma unroll
  for (int mi = 0; mi < 2; ++mi)
#pragma unroll
    for (int ni = 0; ni < 4; ++ni)
#pragma unroll
      for (int r = 0; r < 8; ++r) {
        int grow = m0 + wm + mi * 16 + rowb + r;
        int gcol = n0 + wn + ni * 16 + colb;
        scores[((size_t)z * HW + grow) * HW + gcol] = acc.t[mi][ni][r];
      }
}

__global__ void __launch_bounds__(256)
softmax_kernel(const float* scores, __bf16* probs) {
  const size_t row = blockIdx.x;
  const float* s = scores + row * HW;
  __bf16* p = probs + row * HW;
  __shared__ float red[256];
  float vals[16];
  float mx = -3.4e38f;
#pragma unroll
  for (int i = 0; i < 16; ++i) { vals[i] = s[TID + 256 * i]; mx = fmaxf(mx, vals[i]); }
  red[TID] = mx; __syncthreads();
  for (int st = 128; st > 0; st >>= 1) {
    if (TID < st) red[TID] = fmaxf(red[TID], red[TID + st]);
    __syncthreads();
  }
  mx = red[0]; __syncthreads();
  float sum = 0.f;
#pragma unroll
  for (int i = 0; i < 16; ++i) { vals[i] = __expf(vals[i] - mx); sum += vals[i]; }
  red[TID] = sum; __syncthreads();
  for (int st = 128; st > 0; st >>= 1) {
    if (TID < st) red[TID] += red[TID + st];
    __syncthreads();
  }
  const float inv = 1.0f / red[0];
#pragma unroll
  for (int i = 0; i < 16; ++i) p[TID + 256 * i] = (__bf16)(vals[i] * inv);
}

// AO(512x4096) = V(512x4096) * P^T ; P row-major (i,j) == LDS [n][k] -> direct.
__global__ void __launch_bounds__(256)
gemm_attnout_kernel(const __bf16* qkv, const __bf16* probs, __bf16* ao) {
  __shared__ __bf16 ldsA[2 * ABUF];
  __shared__ __bf16 ldsB[2 * BBUF];
  const int z = blockIdx.z, m0 = blockIdx.y * BM, n0 = blockIdx.x * BN;
  const __bf16* vmat = qkv + (size_t)z * CQKV * HW + (size_t)2 * CCH * HW;
  Acc acc;
  gemm_tile<false>(vmat + (size_t)m0 * HW, HW,
                   probs + ((size_t)z * HW + n0) * HW, HW, HW, ldsA, ldsB, acc);
  EPILOGUE_POS();
#pragma unroll
  for (int mi = 0; mi < 2; ++mi)
#pragma unroll
    for (int ni = 0; ni < 4; ++ni)
#pragma unroll
      for (int r = 0; r < 8; ++r) {
        int grow = m0 + wm + mi * 16 + rowb + r;
        int gcol = n0 + wn + ni * 16 + colb;
        ao[((size_t)z * CCH + grow) * HW + gcol] = (__bf16)acc.t[mi][ni][r];
      }
}

// out = Wout(512x512) * AO(512x4096) + bias + xn (residual), fp32 out
__global__ void __launch_bounds__(256)
gemm_final_kernel(const __bf16* Wout, const __bf16* ao, const float* bias,
                  const float* xn, float* out) {
  __shared__ __bf16 ldsA[2 * ABUF];
  __shared__ __bf16 ldsB[2 * BBUF];
  const int z = blockIdx.z, m0 = blockIdx.y * BM, n0 = blockIdx.x * BN;
  Acc acc;
  gemm_tile<true>(Wout + (size_t)m0 * CCH, CCH,
                  ao + (size_t)z * CCH * HW + n0, HW, CCH, ldsA, ldsB, acc);
  EPILOGUE_POS();
#pragma unroll
  for (int mi = 0; mi < 2; ++mi)
#pragma unroll
    for (int ni = 0; ni < 4; ++ni)
#pragma unroll
      for (int r = 0; r < 8; ++r) {
        int grow = m0 + wm + mi * 16 + rowb + r;
        int gcol = n0 + wn + ni * 16 + colb;
        size_t idx = ((size_t)z * CCH + grow) * HW + gcol;
        out[idx] = acc.t[mi][ni][r] + bias[grow] + xn[idx];
      }
}

// ------------------------------------------------------------------
extern "C" void kernel_launch(void* const* d_in, const int* in_sizes, int n_in,
                              void* d_out, int out_size, void* d_ws,
                              size_t ws_size, hipStream_t stream) {
  (void)in_sizes; (void)n_in; (void)out_size; (void)ws_size;
  const float* x     = (const float*)d_in[0];
  const float* gn_w  = (const float*)d_in[1];
  const float* gn_b  = (const float*)d_in[2];
  const float* qkv_w = (const float*)d_in[3];
  const float* qkv_b = (const float*)d_in[4];
  const float* out_w = (const float*)d_in[5];
  const float* out_b = (const float*)d_in[6];
  float* out = (float*)d_out;

  char* ws = (char*)d_ws;
  size_t off = 0;
  auto alloc = [&](size_t bytes) -> void* {
    void* p = ws + off;
    off += (bytes + 255) & ~(size_t)255;
    return p;
  };
  float*  xn_f32 = (float*) alloc((size_t)BATCH * CCH * HW * 4);
  __bf16* xn_bf  = (__bf16*)alloc((size_t)BATCH * CCH * HW * 2);
  __bf16* wq_bf  = (__bf16*)alloc((size_t)CQKV * CCH * 2);
  __bf16* wo_bf  = (__bf16*)alloc((size_t)CCH * CCH * 2);
  __bf16* qkv_bf = (__bf16*)alloc((size_t)BATCH * CQKV * HW * 2);
  __bf16* qT_bf  = (__bf16*)alloc((size_t)BATCH * HW * CCH * 2);
  float*  scores = (float*) alloc((size_t)BATCH * HW * HW * 4);
  __bf16* probs  = (__bf16*)alloc((size_t)BATCH * HW * HW * 2);
  __bf16* ao_bf  = (__bf16*)alloc((size_t)BATCH * CCH * HW * 2);

  convert_w_kernel<<<(CQKV * CCH + 255) / 256, 256, 0, stream>>>(
      qkv_w, out_w, wq_bf, wo_bf);
  groupnorm_kernel<<<BATCH * NGROUPS, 256, 0, stream>>>(
      x, gn_w, gn_b, xn_f32, xn_bf);
  gemm_qkv_kernel<<<dim3(HW / BN, CQKV / BM, BATCH), 256, 0, stream>>>(
      wq_bf, xn_bf, qkv_b, qkv_bf, qT_bf);
  gemm_scores_kernel<<<dim3(HW / BN, HW / BM, BATCH), 256, 0, stream>>>(
      qT_bf, qkv_bf, scores);
  softmax_kernel<<<BATCH * HW, 256, 0, stream>>>(scores, probs);
  gemm_attnout_kernel<<<dim3(HW / BN, CCH / BM, BATCH), 256, 0, stream>>>(
      qkv_bf, probs, ao_bf);
  gemm_final_kernel<<<dim3(HW / BN, CCH / BM, BATCH), 256, 0, stream>>>(
      wo_bf, ao_bf, out_b, xn_f32, out);
}
